// EntityBranch_34024730919230
// MI455X (gfx1250) — compile-verified, matrix operand order checked by reading, not run on hardware
//
#include <hip/hip_runtime.h>
#include <hip/hip_bf16.h>

// ---------------------------------------------------------------------------
// CDNA5 (gfx1250): wave32, V_WMMA_F32_16X16X32_F16 GEMMs, async global->LDS.
// ---------------------------------------------------------------------------

typedef __attribute__((ext_vector_type(16))) _Float16 v16h;
typedef __attribute__((ext_vector_type(8)))  _Float16 v8h;
typedef __attribute__((ext_vector_type(8)))  float    v8f;
typedef __attribute__((ext_vector_type(4)))  int      v4i;

#define AS1 __attribute__((address_space(1)))
#define AS3 __attribute__((address_space(3)))

#if __has_builtin(__builtin_amdgcn_global_load_async_to_lds_b128) && \
    __has_builtin(__builtin_amdgcn_s_wait_asynccnt)
#define USE_ASYNC_LDS 1
#else
#define USE_ASYNC_LDS 0
#endif

// Problem constants
#define BB   32
#define SS   512
#define HH   768
#define HLH  256
#define G4   1024   // 4*HL
#define EFD  256
#define PDD  128
#define LL   5

// GEMM tiling: 128 threads = 4 waves, 64x64 output tile, K-step 32.
#define TM 64
#define TN 64
#define TK 32
constexpr int A_LD = 40;   // 32 + 8 pad halfs; 80B row stride (16B multiple)
constexpr int B_LD = 40;   // B staged col-major (N-major): Bs[c*B_LD + k]

__device__ __forceinline__ v8f wmma_f16(v16h a, v16h b, v8f c) {
  return __builtin_amdgcn_wmma_f32_16x16x32_f16(
      /*neg_a=*/false, a, /*neg_b=*/false, b,
      /*c_mod=*/(short)0, c, /*reuse_a=*/false, /*reuse_b=*/false);
}

// 16x32 f16 fragment: lanes 0-15 -> row=lane, K halves {k0..k0+7, k0+16..k0+23}
// with k0 = (lane>>4)*8.  Row-contiguous -> two b128 LDS loads.
__device__ __forceinline__ v16h frag_row(const _Float16* base, int row, int ld,
                                         int lane) {
  int r  = row + (lane & 15);
  int kk = (lane >> 4) << 3;
  const _Float16* p = base + r * ld + kk;
  v8h lo = *(const v8h*)(p);
  v8h hi = *(const v8h*)(p + 16);
  v16h a;
#pragma unroll
  for (int i = 0; i < 8; ++i) { a[i] = lo[i]; a[i + 8] = hi[i]; }
  return a;
}

__device__ __forceinline__ float sigm(float x) { return 1.0f / (1.0f + __expf(-x)); }

// ---------------------------------------------------------------------------
// Weight prep: W[K,N] f32 row-major  ->  Wt[N,K] f16 (contiguous along K).
// 32x32 LDS tile transpose; K,N are multiples of 32.  grid.z = weight batch.
// ---------------------------------------------------------------------------
__global__ void __launch_bounds__(256)
transpose_f16(const float* __restrict__ W, _Float16* __restrict__ Wt,
              int K, int N) {
  __shared__ float tile[32][33];
  const float* Wz = W + (long long)blockIdx.z * K * N;
  _Float16* Wtz = Wt + (long long)blockIdx.z * K * N;
  int k0 = blockIdx.y * 32, n0 = blockIdx.x * 32;
  int tx = threadIdx.x & 31, ty = threadIdx.x >> 5;   // 32 x 8
#pragma unroll
  for (int i = ty; i < 32; i += 8)
    tile[i][tx] = Wz[(long long)(k0 + i) * N + (n0 + tx)];
  __syncthreads();
#pragma unroll
  for (int i = ty; i < 32; i += 8)
    Wtz[(long long)(n0 + i) * K + (k0 + tx)] = (_Float16)tile[tx][i];
}

// ---------------------------------------------------------------------------
// Generic tiled GEMM:  C[z] = A[z] @ W[sel(z)] + bias[sel(z)]
//   A  : row-major [M,K] f32 (optional split-K operand A2 for the concat)
//   Wt : f16 [N,K] (pre-transposed), staged to LDS via async-to-LDS b128.
// ---------------------------------------------------------------------------
__global__ void __launch_bounds__(128)
gemm_wmma(const float* __restrict__ A, int lda, long long a_bstride,
          const float* __restrict__ A2, int lda2, int k_split,
          const _Float16* __restrict__ Wt, long long w_bstride,
          const float* __restrict__ bias, long long b_bstride,
          const int* __restrict__ sel,
          float* __restrict__ C, long long c_bstride,
          int M, int N, int K) {
  __shared__ _Float16 As[TM * A_LD];
  __shared__ _Float16 Bs[TN * B_LD];   // Bs[c*B_LD + k]

  const int z = blockIdx.z;
  const int widx = sel ? sel[z] : z;
  const _Float16* Wz = Wt + (long long)widx * w_bstride;
  const float* bz = bias ? (bias + (long long)widx * b_bstride) : nullptr;
  const float* Az = A + (long long)z * a_bstride;
  float* Cz = C + (long long)z * c_bstride;

  const int m_base = blockIdx.y * TM;
  const int n_base = blockIdx.x * TN;
  const int tid  = threadIdx.x;
  const int lane = tid & 31;
  const int wid  = tid >> 5;
  const int wm = wid >> 1;
  const int wn = wid & 1;

  v8f zero = {};
  v8f acc[2][2];
  acc[0][0] = zero; acc[0][1] = zero; acc[1][0] = zero; acc[1][1] = zero;

  for (int k0 = 0; k0 < K; k0 += TK) {
    // ---- B tile: 64 cols x 32 k of f16, contiguous along K. 256 x b128. ----
#if USE_ASYNC_LDS
#pragma unroll
    for (int idx = tid; idx < (TN * TK) / 8; idx += 128) {
      int c  = idx >> 2;
      int kk = (idx & 3) << 3;
      const _Float16* g = Wz + (long long)(n_base + c) * K + (k0 + kk);
      __builtin_amdgcn_global_load_async_to_lds_b128(
          (AS1 v4i*)(unsigned long long)g,
          (AS3 v4i*)&Bs[c * B_LD + kk], 0, 0);
    }
#else
#pragma unroll
    for (int idx = tid; idx < (TN * TK) / 8; idx += 128) {
      int c  = idx >> 2;
      int kk = (idx & 3) << 3;
      const _Float16* g = Wz + (long long)(n_base + c) * K + (k0 + kk);
      *(v8h*)&Bs[c * B_LD + kk] = *(const v8h*)g;
    }
#endif
    // ---- A tile: 64 rows x 32 k, f32 -> f16, 8-wide vector chunks. ----
#pragma unroll
    for (int idx = tid; idx < (TM * TK) / 8; idx += 128) {
      int r  = idx >> 2;
      int kk = (idx & 3) << 3;
      int gr = m_base + r, gk = k0 + kk;
      v8h h8;
      if (gr < M) {
        const float* src = (A2 && gk >= k_split)
                               ? (A2 + (long long)gr * lda2 + (gk - k_split))
                               : (Az + (long long)gr * lda + gk);
        float4 x0 = ((const float4*)src)[0];
        float4 x1 = ((const float4*)src)[1];
        h8[0] = (_Float16)x0.x; h8[1] = (_Float16)x0.y;
        h8[2] = (_Float16)x0.z; h8[3] = (_Float16)x0.w;
        h8[4] = (_Float16)x1.x; h8[5] = (_Float16)x1.y;
        h8[6] = (_Float16)x1.z; h8[7] = (_Float16)x1.w;
      } else {
#pragma unroll
        for (int i = 0; i < 8; ++i) h8[i] = (_Float16)0.0f;
      }
      *(v8h*)&As[r * A_LD + kk] = h8;
    }
    // Prefetch next K tile (global_prefetch_b8).
    if (k0 + TK < K) {
      int gr = m_base + (tid >> 1);
      if (gr < M) __builtin_prefetch(&Az[(long long)gr * lda + k0 + TK], 0, 1);
      __builtin_prefetch(&Wz[(long long)(n_base + (tid >> 1)) * K + k0 + TK], 0, 1);
    }
#if USE_ASYNC_LDS
    __builtin_amdgcn_s_wait_asynccnt(0);
#endif
    __syncthreads();

    v16h a0 = frag_row(As, wm * 32,      A_LD, lane);
    v16h a1 = frag_row(As, wm * 32 + 16, A_LD, lane);
    v16h b0 = frag_row(Bs, wn * 32,      B_LD, lane);
    v16h b1 = frag_row(Bs, wn * 32 + 16, B_LD, lane);
    acc[0][0] = wmma_f16(a0, b0, acc[0][0]);
    acc[0][1] = wmma_f16(a0, b1, acc[0][1]);
    acc[1][0] = wmma_f16(a1, b0, acc[1][0]);
    acc[1][1] = wmma_f16(a1, b1, acc[1][1]);
    __syncthreads();
  }

  // Store (C layout: lane = N col within tile, VGPR r = M row, +8 hi lanes).
#pragma unroll
  for (int mi = 0; mi < 2; ++mi)
#pragma unroll
    for (int ni = 0; ni < 2; ++ni) {
      int col  = n_base + wn * 32 + ni * 16 + (lane & 15);
      int mrow = m_base + wm * 32 + mi * 16 + ((lane >> 4) << 3);
      float bv = bz ? bz[col] : 0.0f;
#pragma unroll
      for (int r = 0; r < 8; ++r) {
        int m = mrow + r;
        if (m < M) Cz[(long long)m * N + col] = acc[mi][ni][r] + bv;
      }
    }
}

// ---------------------------------------------------------------------------
// Row LayerNorm + ReLU (in place).  One wave per row, 8 rows per block.
// ---------------------------------------------------------------------------
__global__ void __launch_bounds__(256)
ln_relu(float* __restrict__ X, int rows, int D,
        const float* __restrict__ g, const float* __restrict__ bt,
        const int* __restrict__ sel, int rows_per_sel, int sel_stride) {
  int row  = blockIdx.x * 8 + (threadIdx.x >> 5);
  int lane = threadIdx.x & 31;
  if (row >= rows) return;
  const float* gg = g;
  const float* bb = bt;
  if (sel) {
    int s = sel[row / rows_per_sel];
    gg += (long long)s * sel_stride;
    bb += (long long)s * sel_stride;
  }
  float* x = X + (long long)row * D;
  float s1 = 0.0f, s2 = 0.0f;
  for (int i = lane; i < D; i += 32) { float v = x[i]; s1 += v; s2 += v * v; }
#pragma unroll
  for (int m = 16; m > 0; m >>= 1) {
    s1 += __shfl_xor(s1, m, 32);
    s2 += __shfl_xor(s2, m, 32);
  }
  float mean = s1 / D;
  float var  = s2 / D - mean * mean;
  float rs   = rsqrtf(var + 1e-5f);
  for (int i = lane; i < D; i += 32) {
    float v = (x[i] - mean) * rs * gg[i] + bb[i];
    x[i] = v > 0.0f ? v : 0.0f;
  }
}

// ---------------------------------------------------------------------------
// LSTM recurrence with WMMA.  grid(4): dir = bx&1, batch-half = bx>>1.
// 16 batches (one M tile), N=1024 gates, K=256.  WhhT = f16 [1024,256],
// kept register-resident (16 fragments/wave); h/c state in LDS.
// ---------------------------------------------------------------------------
#define LSTM_SMEM (16 * G4 * 4 + 16 * HLH * 4 + 16 * HLH * 2)

__global__ void __launch_bounds__(1024)
lstm_wmma(const float* __restrict__ xpf, const float* __restrict__ xpb,
          const _Float16* __restrict__ whhtf, const _Float16* __restrict__ whhtb,
          float* __restrict__ hf, float* __restrict__ hb) {
  extern __shared__ char smem[];
  float*    g_s = (float*)smem;                                   // 16 x 1024
  float*    c_s = (float*)(smem + 16 * G4 * 4);                   // 16 x 256
  _Float16* h_s = (_Float16*)(smem + 16 * G4 * 4 + 16 * HLH * 4); // 16 x 256

  const int dir  = blockIdx.x & 1;
  const int boff = (blockIdx.x >> 1) * 16;
  const float*    xp   = dir ? xpb : xpf;
  const _Float16* WhhT = dir ? whhtb : whhtf;
  float*          hout = dir ? hb : hf;

  const int tid  = threadIdx.x;
  const int lane = tid & 31;
  const int wv   = tid >> 5;       // 0..31 -> 32 gate-columns each
  const int ncol0 = wv * 32;

  for (int i = tid; i < 16 * HLH; i += 1024) {
    c_s[i] = 0.0f;
    h_s[i] = (_Float16)0.0f;
  }

  // Preload B panel: WhhT[n, k] contiguous along k -> b128 loads.
  v16h Bf[8][2];
#pragma unroll
  for (int kt = 0; kt < 8; ++kt)
#pragma unroll
    for (int ni = 0; ni < 2; ++ni) {
      int c  = ncol0 + ni * 16 + (lane & 15);
      int kb = kt * 32 + ((lane >> 4) << 3);
      const _Float16* p = WhhT + (long long)c * HLH + kb;
      v8h lo = *(const v8h*)(p);
      v8h hi = *(const v8h*)(p + 16);
#pragma unroll
      for (int i = 0; i < 8; ++i) {
        Bf[kt][ni][i]     = lo[i];
        Bf[kt][ni][i + 8] = hi[i];
      }
    }
  __syncthreads();

  v8f zero = {};
  for (int step = 0; step < SS; ++step) {
    const int t = dir ? (SS - 1 - step) : step;
    v8f acc0 = zero, acc1 = zero;
#pragma unroll
    for (int kt = 0; kt < 8; ++kt) {
      v16h a;
      {
        int r  = lane & 15;
        int kk = kt * 32 + ((lane >> 4) << 3);
        const _Float16* p = h_s + r * HLH + kk;
        v8h lo = *(const v8h*)(p);
        v8h hi = *(const v8h*)(p + 16);
#pragma unroll
        for (int i = 0; i < 8; ++i) { a[i] = lo[i]; a[i + 8] = hi[i]; }
      }
      acc0 = wmma_f16(a, Bf[kt][0], acc0);
      acc1 = wmma_f16(a, Bf[kt][1], acc1);
    }
    // g = h@Whh + xp[:, t, :]
#pragma unroll
    for (int ni = 0; ni < 2; ++ni) {
      v8f ac = ni ? acc1 : acc0;
      int col = ncol0 + ni * 16 + (lane & 15);
      int bl0 = (lane >> 4) << 3;
#pragma unroll
      for (int r = 0; r < 8; ++r) {
        int bl = bl0 + r;
        g_s[bl * G4 + col] =
            ac[r] + xp[((long long)(boff + bl) * SS + t) * G4 + col];
      }
    }
    __syncthreads();
    // Elementwise gates: 16*256 = 4096 cells, 4 per thread.
#pragma unroll
    for (int p = 0; p < 4; ++p) {
      int e = tid + p * 1024;
      int bl = e >> 8, j = e & 255;
      float gi = g_s[bl * G4 + j];
      float gf = g_s[bl * G4 + 256 + j];
      float gg = g_s[bl * G4 + 512 + j];
      float go = g_s[bl * G4 + 768 + j];
      float c  = sigm(gf) * c_s[e] + sigm(gi) * tanhf(gg);
      float h  = sigm(go) * tanhf(c);
      c_s[e] = c;
      h_s[e] = (_Float16)h;
      hout[((long long)(boff + bl) * SS + t) * HLH + j] = h;
    }
    __syncthreads();
  }
}

// ---------------------------------------------------------------------------
// Emissions: em[row, l] = -||q[row] - sp[l]||  (q: [rows,128], sp: [5,128])
// ---------------------------------------------------------------------------
__global__ void __launch_bounds__(256)
emissions_kernel(const float* __restrict__ q, const float* __restrict__ sp,
                 float* __restrict__ em, int rows) {
  __shared__ float sps[LL * PDD];
  for (int i = threadIdx.x; i < LL * PDD; i += 256) sps[i] = sp[i];
  __syncthreads();
  int row = blockIdx.x * 256 + threadIdx.x;
  if (row >= rows) return;
  const float* qr = q + (long long)row * PDD;
#pragma unroll
  for (int l = 0; l < LL; ++l) {
    float d2 = 0.0f;
    for (int i = 0; i < PDD; ++i) {
      float d = qr[i] - sps[l * PDD + i];
      d2 += d * d;
    }
    em[(long long)row * LL + l] = -sqrtf(fmaxf(d2, 0.0f));
  }
}

// ---------------------------------------------------------------------------
// Prototype loss:  pl = -mean(log_softmax(-cdist(sp,protos)/T)[a, lab[a]])
// ---------------------------------------------------------------------------
__global__ void proto_loss_kernel(const float* __restrict__ sp,
                                  const float* __restrict__ protos,
                                  const float* __restrict__ temp,
                                  const int* __restrict__ slab,
                                  float* __restrict__ pl) {
  if (threadIdx.x != 0 || blockIdx.x != 0) return;
  float total = 0.0f;
  float invT = 1.0f / temp[0];
  for (int a = 0; a < LL; ++a) {
    float logits[LL];
    for (int l = 0; l < LL; ++l) {
      float d2 = 0.0f;
      for (int i = 0; i < PDD; ++i) {
        float d = sp[a * PDD + i] - protos[l * PDD + i];
        d2 += d * d;
      }
      logits[l] = -sqrtf(fmaxf(d2, 0.0f)) * invT;
    }
    float mx = logits[0];
    for (int l = 1; l < LL; ++l) mx = fmaxf(mx, logits[l]);
    float se = 0.0f;
    for (int l = 0; l < LL; ++l) se += __expf(logits[l] - mx);
    float lse = mx + __logf(se);
    total += -(logits[slab[a]] - lse);
  }
  pl[0] = total / (float)LL;
}

// ---------------------------------------------------------------------------
// CRF NLL: 1 wave, one batch chain per lane; combine with proto loss.
// ---------------------------------------------------------------------------
__global__ void crf_kernel(const float* __restrict__ em,
                           const int* __restrict__ tags,
                           const float* __restrict__ start,
                           const float* __restrict__ end,
                           const float* __restrict__ trans,
                           const float* __restrict__ pl,
                           float* __restrict__ out) {
  int b = threadIdx.x;   // 0..31
  const float* e  = em + (long long)b * SS * LL;
  const int*   tg = tags + b * SS;
  float tr[LL][LL], st[LL], en[LL];
#pragma unroll
  for (int i = 0; i < LL; ++i) {
    st[i] = start[i];
    en[i] = end[i];
#pragma unroll
    for (int j = 0; j < LL; ++j) tr[i][j] = trans[i * LL + j];
  }
  int prev = tg[0];
  float num = st[prev] + e[prev];
  for (int t = 1; t < SS; ++t) {
    int cur = tg[t];
    num += tr[prev][cur] + e[t * LL + cur];
    prev = cur;
  }
  num += en[prev];
  float alpha[LL];
#pragma unroll
  for (int j = 0; j < LL; ++j) alpha[j] = st[j] + e[j];
  for (int t = 1; t < SS; ++t) {
    float na[LL];
#pragma unroll
    for (int j = 0; j < LL; ++j) {
      float mx = alpha[0] + tr[0][j];
#pragma unroll
      for (int i = 1; i < LL; ++i) mx = fmaxf(mx, alpha[i] + tr[i][j]);
      float se = 0.0f;
#pragma unroll
      for (int i = 0; i < LL; ++i) se += __expf(alpha[i] + tr[i][j] - mx);
      na[j] = mx + __logf(se) + e[t * LL + j];
    }
#pragma unroll
    for (int j = 0; j < LL; ++j) alpha[j] = na[j];
  }
  float mx = alpha[0] + en[0];
#pragma unroll
  for (int j = 1; j < LL; ++j) mx = fmaxf(mx, alpha[j] + en[j]);
  float se = 0.0f;
#pragma unroll
  for (int j = 0; j < LL; ++j) se += __expf(alpha[j] + en[j] - mx);
  float logZ = mx + __logf(se);
  float v = -(num - logZ);
#pragma unroll
  for (int m = 16; m > 0; m >>= 1) v += __shfl_xor(v, m, 32);
  if (b == 0) out[0] = v / (float)BB + 0.5f * pl[0];
}

// ---------------------------------------------------------------------------
// Host-side orchestration
// ---------------------------------------------------------------------------
extern "C" void kernel_launch(void* const* d_in, const int* in_sizes, int n_in,
                              void* d_out, int out_size, void* d_ws,
                              size_t ws_size, hipStream_t stream) {
  (void)in_sizes; (void)n_in; (void)out_size; (void)ws_size;

  const float* x      = (const float*)d_in[0];
  const int*   langs  = (const int*)d_in[2];
  const int*   labels = (const int*)d_in[3];
  const float* sef    = (const float*)d_in[4];
  const int*   slab   = (const int*)d_in[5];
  const float* aW1    = (const float*)d_in[6];
  const float* ab1    = (const float*)d_in[7];
  const float* alng   = (const float*)d_in[8];
  const float* alnb   = (const float*)d_in[9];
  const float* aW2    = (const float*)d_in[10];
  const float* ab2    = (const float*)d_in[11];
  const float* Wih_f  = (const float*)d_in[12];
  const float* Whh_f  = (const float*)d_in[13];
  const float* b_f    = (const float*)d_in[14];
  const float* Wih_b  = (const float*)d_in[15];
  const float* Whh_b  = (const float*)d_in[16];
  const float* b_b    = (const float*)d_in[17];
  const float* projW  = (const float*)d_in[18];
  const float* projb  = (const float*)d_in[19];
  const float* pW1    = (const float*)d_in[20];
  const float* pb1    = (const float*)d_in[21];
  const float* plng   = (const float*)d_in[22];
  const float* plnb   = (const float*)d_in[23];
  const float* pW2    = (const float*)d_in[24];
  const float* pb2    = (const float*)d_in[25];
  const float* protos = (const float*)d_in[26];
  const float* temp   = (const float*)d_in[27];
  const float* startt = (const float*)d_in[28];
  const float* endt   = (const float*)d_in[29];
  const float* trans  = (const float*)d_in[30];

  const long long NT = (long long)BB * SS;   // 16384 token rows

  size_t off = 0;
  auto carve = [&](size_t bytes) -> void* {
    void* p = (char*)d_ws + off;
    off += (bytes + 255) & ~(size_t)255;
    return p;
  };
  // f32 activations
  float*    h1    = (float*)carve((size_t)NT * HH * 4);
  float*    adap  = (float*)carve((size_t)NT * HH * 4);
  float*    xpf   = (float*)carve((size_t)NT * G4 * 4);
  float*    xpb   = (float*)carve((size_t)NT * G4 * 4);
  float*    hf    = (float*)carve((size_t)NT * HLH * 4);
  float*    hb    = (float*)carve((size_t)NT * HLH * 4);
  float*    ent   = (float*)carve((size_t)NT * EFD * 4);
  float*    ph    = (float*)carve((size_t)NT * PDD * 4);
  float*    pfeat = (float*)carve((size_t)NT * PDD * 4);
  float*    sph   = (float*)carve((size_t)LL * PDD * 4);
  float*    sp    = (float*)carve((size_t)LL * PDD * 4);
  float*    em    = (float*)carve((size_t)NT * LL * 4);
  float*    pl    = (float*)carve(256);
  // f16 pre-transposed weights [N,K]
  _Float16* aW1t  = (_Float16*)carve((size_t)5 * HH * HH * 2);
  _Float16* aW2t  = (_Float16*)carve((size_t)5 * HH * HH * 2);
  _Float16* wihft = (_Float16*)carve((size_t)HH * G4 * 2);
  _Float16* wihbt = (_Float16*)carve((size_t)HH * G4 * 2);
  _Float16* whhft = (_Float16*)carve((size_t)HLH * G4 * 2);
  _Float16* whhbt = (_Float16*)carve((size_t)HLH * G4 * 2);
  _Float16* projt = (_Float16*)carve((size_t)2 * HLH * EFD * 2);
  _Float16* pW1t  = (_Float16*)carve((size_t)EFD * PDD * 2);
  _Float16* pW2t  = (_Float16*)carve((size_t)PDD * PDD * 2);

  (void)hipFuncSetAttribute((const void*)lstm_wmma,
                            hipFuncAttributeMaxDynamicSharedMemorySize,
                            LSTM_SMEM);

  // 0) One-time weight transposes to f16 [N,K]
  transpose_f16<<<dim3(HH / 32, HH / 32, 5), 256, 0, stream>>>(aW1, aW1t, HH, HH);
  transpose_f16<<<dim3(HH / 32, HH / 32, 5), 256, 0, stream>>>(aW2, aW2t, HH, HH);
  transpose_f16<<<dim3(G4 / 32, HH / 32, 1), 256, 0, stream>>>(Wih_f, wihft, HH, G4);
  transpose_f16<<<dim3(G4 / 32, HH / 32, 1), 256, 0, stream>>>(Wih_b, wihbt, HH, G4);
  transpose_f16<<<dim3(G4 / 32, HLH / 32, 1), 256, 0, stream>>>(Whh_f, whhft, HLH, G4);
  transpose_f16<<<dim3(G4 / 32, HLH / 32, 1), 256, 0, stream>>>(Whh_b, whhbt, HLH, G4);
  transpose_f16<<<dim3(EFD / 32, (2 * HLH) / 32, 1), 256, 0, stream>>>(projW, projt, 2 * HLH, EFD);
  transpose_f16<<<dim3(PDD / 32, EFD / 32, 1), 256, 0, stream>>>(pW1, pW1t, EFD, PDD);
  transpose_f16<<<dim3(PDD / 32, PDD / 32, 1), 256, 0, stream>>>(pW2, pW2t, PDD, PDD);

  // 1) h1 = x @ aW1[lang] + ab1[lang]
  gemm_wmma<<<dim3(HH / TN, SS / TM, BB), 128, 0, stream>>>(
      x, HH, (long long)SS * HH, nullptr, 0, 0,
      aW1t, (long long)HH * HH, ab1, HH, langs,
      h1, (long long)SS * HH, SS, HH, HH);

  // 2) LN(alng[lang], alnb[lang]) + ReLU
  ln_relu<<<dim3((int)(NT / 8)), 256, 0, stream>>>(
      h1, (int)NT, HH, alng, alnb, langs, SS, HH);

  // 3) adapted = h1 @ aW2[lang] + ab2[lang]
  gemm_wmma<<<dim3(HH / TN, SS / TM, BB), 128, 0, stream>>>(
      h1, HH, (long long)SS * HH, nullptr, 0, 0,
      aW2t, (long long)HH * HH, ab2, HH, langs,
      adap, (long long)SS * HH, SS, HH, HH);

  // 4/5) LSTM input projections
  gemm_wmma<<<dim3(G4 / TN, (int)(NT / TM), 1), 128, 0, stream>>>(
      adap, HH, 0, nullptr, 0, 0, wihft, 0, b_f, 0, nullptr,
      xpf, 0, (int)NT, G4, HH);
  gemm_wmma<<<dim3(G4 / TN, (int)(NT / TM), 1), 128, 0, stream>>>(
      adap, HH, 0, nullptr, 0, 0, wihbt, 0, b_b, 0, nullptr,
      xpb, 0, (int)NT, G4, HH);

  // 7) Recurrence: 2 dirs x 2 batch-halves = 4 workgroups
  lstm_wmma<<<dim3(4), 1024, LSTM_SMEM, stream>>>(xpf, xpb, whhft, whhbt, hf, hb);

  // 8) entity_features = concat(hf, hb) @ projW + projb
  gemm_wmma<<<dim3(EFD / TN, (int)(NT / TM), 1), 128, 0, stream>>>(
      hf, HLH, 0, hb, HLH, HLH, projt, 0, projb, 0, nullptr,
      ent, 0, (int)NT, EFD, 2 * HLH);

  // 9-11) proto projection (query path)
  gemm_wmma<<<dim3(PDD / TN, (int)(NT / TM), 1), 128, 0, stream>>>(
      ent, EFD, 0, nullptr, 0, 0, pW1t, 0, pb1, 0, nullptr,
      ph, 0, (int)NT, PDD, EFD);
  ln_relu<<<dim3((int)(NT / 8)), 256, 0, stream>>>(
      ph, (int)NT, PDD, plng, plnb, nullptr, 1, 0);
  gemm_wmma<<<dim3(PDD / TN, (int)(NT / TM), 1), 128, 0, stream>>>(
      ph, PDD, 0, nullptr, 0, 0, pW2t, 0, pb2, 0, nullptr,
      pfeat, 0, (int)NT, PDD, PDD);

  // 12-14) proto projection (support path, M=5 via row guards)
  gemm_wmma<<<dim3(PDD / TN, 1, 1), 128, 0, stream>>>(
      sef, EFD, 0, nullptr, 0, 0, pW1t, 0, pb1, 0, nullptr,
      sph, 0, LL, PDD, EFD);
  ln_relu<<<dim3(1), 256, 0, stream>>>(sph, LL, PDD, plng, plnb, nullptr, 1, 0);
  gemm_wmma<<<dim3(PDD / TN, 1, 1), 128, 0, stream>>>(
      sph, PDD, 0, nullptr, 0, 0, pW2t, 0, pb2, 0, nullptr,
      sp, 0, LL, PDD, PDD);

  // 15) emissions
  emissions_kernel<<<dim3((int)((NT + 255) / 256)), 256, 0, stream>>>(
      pfeat, sp, em, (int)NT);

  // 16) prototype loss scalar
  proto_loss_kernel<<<dim3(1), 32, 0, stream>>>(sp, protos, temp, slab, pl);

  // 17) CRF NLL + combine -> d_out[0]
  crf_kernel<<<dim3(1), 32, 0, stream>>>(em, labels, startt, endt, trans, pl,
                                         (float*)d_out);
}
